// VisionMamba_29231547417011
// MI455X (gfx1250) — matrix operand order repaired and checked by reading, not compiled, not run on hardware
//
#include <hip/hip_runtime.h>
#include <math.h>

#define B_SZ   64
#define LSEQ   196
#define DM     192
#define DI     384
#define NSTATE 16
#define DCONV  4
#define DTR    12
#define NCLS   1000
#define DEPTH  24
#define KPATCH 768          // C_IN * 16 * 16
#define ML     (B_SZ * LSEQ) // 12544

typedef unsigned short ushort_t;
typedef __bf16 bf16_t;
typedef bf16_t v16bf __attribute__((ext_vector_type(16)));
typedef float  v8f   __attribute__((ext_vector_type(8)));

// Pointer types for the CDNA5 async global->LDS copy builtin:
// param 1 is a global (AS1) v4i32*, param 2 the LDS (AS3) side.
typedef int vsint4 __attribute__((vector_size(16)));
typedef __attribute__((address_space(1))) vsint4* as1_v4i;
typedef __attribute__((address_space(3))) vsint4* as3_v4i;

#if defined(__has_builtin)
#  if __has_builtin(__builtin_amdgcn_global_load_async_to_lds_b128)
#    define HAVE_ASYNC_LDS 1
#  endif
#endif
#ifndef HAVE_ASYNC_LDS
#  define HAVE_ASYNC_LDS 0
#endif

#if defined(__has_builtin)
#  if __has_builtin(__builtin_amdgcn_s_wait_asynccnt)
#    define WAIT_ASYNC(n) __builtin_amdgcn_s_wait_asynccnt(n)
#  endif
#endif
#ifndef WAIT_ASYNC
#  define WAIT_ASYNC(n) asm volatile("s_wait_asynccnt %0" :: "i"(n) : "memory")
#endif

__device__ __forceinline__ ushort_t f2bf(float f) {
  unsigned u = __float_as_uint(f);
  u += 0x7FFFu + ((u >> 16) & 1u);          // round-to-nearest-even
  return (ushort_t)(u >> 16);
}
__device__ __forceinline__ float bf2f(ushort_t h) {
  return __uint_as_float(((unsigned)h) << 16);
}

// ---------------- elementwise f32 -> bf16 ----------------
__global__ void k_f32_to_bf16(const float* __restrict__ src, ushort_t* __restrict__ dst, int n) {
  for (int i = blockIdx.x * blockDim.x + threadIdx.x; i < n; i += gridDim.x * blockDim.x)
    dst[i] = f2bf(src[i]);
}

// ---------------- im2col for patch embedding ----------------
__global__ void k_im2col(const float* __restrict__ x, ushort_t* __restrict__ a) {
  int idx = blockIdx.x * blockDim.x + threadIdx.x;
  if (idx >= ML * KPATCH) return;
  int k = idx % KPATCH, row = idx / KPATCH;
  int b = row / LSEQ, l = row % LSEQ;
  int c = k >> 8, rem = k & 255, iy = rem >> 4, ix = rem & 15;
  int py = l / 14, px = l % 14;
  float v = x[(((size_t)b * 3 + c) * 224 + (py * 16 + iy)) * 224 + (px * 16 + ix)];
  a[idx] = f2bf(v);
}

// ---------------- WMMA GEMM: C[M,N] = A[M,K] * W[N,K]^T (+bias) ----------------
// A, W bf16 row-major; C f32. Block = 128 thr (4 waves), tile 64x64, K-step 32.
// Double-buffered LDS, async global->LDS staging pipelined against WMMA.
// Out-of-range rows are *clamped* (not masked): garbage rows/cols of D are
// simply never stored, so every wave issues exactly 4 asyncs per K-step and
// "s_wait_asynccnt 4" releases exactly the previous buffer's copies.
#define LDSTRIDE 40
__global__ __launch_bounds__(128) void k_wmma_gemm(
    const ushort_t* __restrict__ A, const ushort_t* __restrict__ W,
    float* __restrict__ C, int M, int N, int K, const float* __restrict__ bias)
{
  __shared__ ushort_t lds_a[2][64][LDSTRIDE];
  __shared__ ushort_t lds_w[2][64][LDSTRIDE];
  const int tid  = threadIdx.x;
  const int wave = tid >> 5, lane = tid & 31;
  const int m16  = lane & 15, half = lane >> 4;
  const int mt = blockIdx.y * 64, nt = blockIdx.x * 64;

  v8f acc[4];
  #pragma unroll
  for (int t = 0; t < 4; ++t)
    #pragma unroll
    for (int r = 0; r < 8; ++r) acc[t][r] = 0.f;

  const int arow = tid >> 1;        // 0..63
  const int acol = (tid & 1) * 16;  // 0 or 16
  const int gm = min(mt + arow, M - 1);   // clamp: OOB rows are staged but never stored
  const int gn = min(nt + arow, N - 1);

  const ushort_t* gA = A + (size_t)gm * K + acol;
  const ushort_t* gW = W + (size_t)gn * K + acol;

#if HAVE_ASYNC_LDS
  #define STAGE_TILE(buf, k0)                                                     \
    do {                                                                          \
      __builtin_amdgcn_global_load_async_to_lds_b128(                             \
          (as1_v4i)(gA + (k0)), (as3_v4i)&lds_a[buf][arow][acol], 0, 0);          \
      __builtin_amdgcn_global_load_async_to_lds_b128(                             \
          (as1_v4i)(gA + (k0)), (as3_v4i)&lds_a[buf][arow][acol], 16, 0);         \
      __builtin_amdgcn_global_load_async_to_lds_b128(                             \
          (as1_v4i)(gW + (k0)), (as3_v4i)&lds_w[buf][arow][acol], 0, 0);          \
      __builtin_amdgcn_global_load_async_to_lds_b128(                             \
          (as1_v4i)(gW + (k0)), (as3_v4i)&lds_w[buf][arow][acol], 16, 0);         \
    } while (0)
#else
  #define STAGE_TILE(buf, k0)                                                     \
    do {                                                                          \
      union { uint4 q[2]; ushort_t us[16]; } pa, pw;                              \
      const uint4* sa = reinterpret_cast<const uint4*>(gA + (k0));                \
      const uint4* sw = reinterpret_cast<const uint4*>(gW + (k0));                \
      pa.q[0] = sa[0]; pa.q[1] = sa[1];                                           \
      pw.q[0] = sw[0]; pw.q[1] = sw[1];                                           \
      for (int e = 0; e < 16; ++e) lds_a[buf][arow][acol + e] = pa.us[e];         \
      for (int e = 0; e < 16; ++e) lds_w[buf][arow][acol + e] = pw.us[e];         \
    } while (0)
#endif

  // Prologue: stage first tile into buffer 0.
  STAGE_TILE(0, 0);

  const int nk = K >> 5;
  for (int it = 0; it < nk; ++it) {
    const int cur = it & 1;
#if HAVE_ASYNC_LDS
    if (it + 1 < nk) {
      STAGE_TILE(cur ^ 1, (it + 1) << 5);  // next tile in flight during compute
      WAIT_ASYNC(4);                       // oldest 4 = current buffer done
    } else {
      WAIT_ASYNC(0);                       // drain
    }
#else
    // fallback: synchronous staging of the current tile happened in the
    // previous iteration (or the prologue); stage the next tile into the other buffer now.
    if (it + 1 < nk) STAGE_TILE(cur ^ 1, (it + 1) << 5);
#endif
    __syncthreads();  // all waves' copies into `cur` visible

    // A fragment: 16x32 bf16 (ISA 7.12.2): lane half selects K lo/hi 8-blocks.
    // Both 16B chunks are 16B-aligned in LDS -> ds_load_b128.
    union { v16bf v; uint4 q[2]; } af;
    {
      const uint4* pa = reinterpret_cast<const uint4*>(&lds_a[cur][wave * 16 + m16][half * 8]);
      af.q[0] = pa[0];   // K = half*8 .. half*8+7
      af.q[1] = pa[2];   // K = 16+half*8 .. +7  (+32 bytes)
    }
    #pragma unroll
    for (int t = 0; t < 4; ++t) {
      // B fragment: 32x16, lane = N col, lane-half selects K lo/hi 8-blocks (same as A).
      union { v16bf v; uint4 q[2]; } bfm;
      const uint4* pb = reinterpret_cast<const uint4*>(&lds_w[cur][t * 16 + m16][half * 8]);
      bfm.q[0] = pb[0];
      bfm.q[1] = pb[2];
      acc[t] = __builtin_amdgcn_wmma_f32_16x16x32_bf16(
          false, af.v, false, bfm.v, (short)0, acc[t], false, false);
    }
    __syncthreads();  // done reading `cur` before it is re-staged
  }
  #undef STAGE_TILE

  // D fragment layout: lane col = lane&15, rows = half*8 + r
  #pragma unroll
  for (int t = 0; t < 4; ++t) {
    int col = nt + t * 16 + m16;
    #pragma unroll
    for (int r = 0; r < 8; ++r) {
      int row = mt + wave * 16 + half * 8 + r;
      if (row < M && col < N) {
        float v = acc[t][r];
        if (bias) v += bias[col];
        C[(size_t)row * N + col] = v;
      }
    }
  }
}

// ---------------- residual add + LayerNorm -> bf16 ----------------
__global__ __launch_bounds__(256) void k_resln(
    float* __restrict__ resid, const float* __restrict__ hidden, int first,
    const float* __restrict__ w, const float* __restrict__ b, ushort_t* __restrict__ hn)
{
  int wave = threadIdx.x >> 5, lane = threadIdx.x & 31;
  int row = blockIdx.x * 8 + wave;
  if (row >= ML) return;
  const float* hr = hidden + (size_t)row * DM;
  float* rr = resid + (size_t)row * DM;
  float v[6], s = 0.f;
  #pragma unroll
  for (int j = 0; j < 6; ++j) {
    int idx = lane + 32 * j;
    float r = first ? hr[idx] : (rr[idx] + hr[idx]);
    rr[idx] = r; v[j] = r; s += r;
  }
  #pragma unroll
  for (int off = 16; off >= 1; off >>= 1) s += __shfl_xor(s, off, 32);
  float mu = s * (1.f / DM);
  float q = 0.f;
  #pragma unroll
  for (int j = 0; j < 6; ++j) { float d = v[j] - mu; q += d * d; }
  #pragma unroll
  for (int off = 16; off >= 1; off >>= 1) q += __shfl_xor(q, off, 32);
  float rs = rsqrtf(q * (1.f / DM) + 1e-5f);
  #pragma unroll
  for (int j = 0; j < 6; ++j) {
    int idx = lane + 32 * j;
    hn[(size_t)row * DM + idx] = f2bf((v[j] - mu) * rs * w[idx] + b[idx]);
  }
}

// ---------------- depthwise causal conv (k=4) + SiLU -> bf16 ----------------
__global__ void k_conv_silu(const float* __restrict__ xz, const float* __restrict__ cw,
                            const float* __restrict__ cb, ushort_t* __restrict__ xc)
{
  int idx = blockIdx.x * blockDim.x + threadIdx.x;
  if (idx >= ML * DI) return;
  int d = idx % DI, bl = idx / DI;
  int l = bl % LSEQ, b = bl / LSEQ;
  float acc = cb[d];
  #pragma unroll
  for (int j = 0; j < DCONV; ++j) {
    int ls = l - 3 + j;
    if (ls >= 0) acc += cw[d * DCONV + j] * xz[((size_t)(b * LSEQ + ls)) * 768 + d];
  }
  float s = acc * (1.f / (1.f + __expf(-acc)));
  xc[idx] = f2bf(s);
}

// ---------------- dt projection + softplus ----------------
__global__ void k_dtproj(const float* __restrict__ xdbl, const float* __restrict__ dtw,
                         const float* __restrict__ dtb, float* __restrict__ dt)
{
  int idx = blockIdx.x * blockDim.x + threadIdx.x;
  if (idx >= ML * DI) return;
  int d = idx % DI, bl = idx / DI;
  float acc = dtb[d];
  const float* xr = xdbl + (size_t)bl * 44;
  #pragma unroll
  for (int r = 0; r < DTR; ++r) acc += xr[r] * dtw[d * DTR + r];
  dt[idx] = (acc > 20.f) ? acc : log1pf(__expf(acc));
}

// ---------------- selective scan (+ D skip + SiLU(z) gate) -> bf16 ----------------
__global__ __launch_bounds__(384) void k_scan(
    const ushort_t* __restrict__ xc, const float* __restrict__ dt,
    const float* __restrict__ xdbl, const float* __restrict__ A_log,
    const float* __restrict__ Dskip, const float* __restrict__ xz,
    ushort_t* __restrict__ ybf)
{
  __shared__ float sB[NSTATE], sC[NSTATE];
  int b = blockIdx.x, d = threadIdx.x;
  float Aa[NSTATE];
  #pragma unroll
  for (int n = 0; n < NSTATE; ++n) Aa[n] = -__expf(A_log[d * NSTATE + n]);
  float Ds = Dskip[d];
  float h[NSTATE];
  #pragma unroll
  for (int n = 0; n < NSTATE; ++n) h[n] = 0.f;

  for (int l = 0; l < LSEQ; ++l) {
    size_t bl = (size_t)b * LSEQ + l;
    if (d < 32) {
      float v = xdbl[bl * 44 + DTR + d];
      if (d < NSTATE) sB[d] = v; else sC[d - NSTATE] = v;
    }
    __syncthreads();
    float dtv = dt[bl * DI + d];
    float u   = bf2f(xc[bl * DI + d]);
    float du  = dtv * u;
    float y = 0.f;
    #pragma unroll
    for (int n = 0; n < NSTATE; ++n) {
      h[n] = __expf(dtv * Aa[n]) * h[n] + du * sB[n];
      y += h[n] * sC[n];
    }
    y += u * Ds;
    float z = xz[bl * 768 + DI + d];
    float sz = z * (1.f / (1.f + __expf(-z)));
    ybf[bl * DI + d] = f2bf(y * sz);
    __syncthreads();
  }
}

// ---------------- gather last-token rows for head ----------------
__global__ void k_pool(const ushort_t* __restrict__ hn, ushort_t* __restrict__ pooled) {
  int idx = blockIdx.x * blockDim.x + threadIdx.x;
  if (idx >= B_SZ * DM) return;
  int b = idx / DM, c = idx % DM;
  pooled[idx] = hn[((size_t)b * LSEQ + (LSEQ - 1)) * DM + c];
}

// ---------------- host orchestration ----------------
extern "C" void kernel_launch(void* const* d_in, const int* in_sizes, int n_in,
                              void* d_out, int out_size, void* d_ws, size_t ws_size,
                              hipStream_t stream)
{
  (void)in_sizes; (void)n_in; (void)out_size; (void)ws_size;
  const float* x         = (const float*)d_in[0];
  const float* patch_w   = (const float*)d_in[1];
  const float* patch_b   = (const float*)d_in[2];
  const float* in_proj_w = (const float*)d_in[3];
  const float* conv_w    = (const float*)d_in[4];
  const float* conv_b    = (const float*)d_in[5];
  const float* x_proj_w  = (const float*)d_in[6];
  const float* dt_proj_w = (const float*)d_in[7];
  const float* dt_proj_b = (const float*)d_in[8];
  const float* A_log     = (const float*)d_in[9];
  const float* D_skip    = (const float*)d_in[10];
  const float* out_proj_w= (const float*)d_in[11];
  const float* norm_w    = (const float*)d_in[12];
  const float* norm_b    = (const float*)d_in[13];
  const float* normf_w   = (const float*)d_in[14];
  const float* normf_b   = (const float*)d_in[15];
  const float* head_w    = (const float*)d_in[16];
  const float* head_b    = (const float*)d_in[17];
  float* out = (float*)d_out;

  char* ws = (char*)d_ws;
  size_t off = 0;
  auto alloc = [&](size_t bytes) -> char* {
    char* p = ws + off; off += (bytes + 255) & ~(size_t)255; return p;
  };
  ushort_t* bfPatch   = (ushort_t*)alloc((size_t)DM * KPATCH * 2);
  ushort_t* bfInproj  = (ushort_t*)alloc((size_t)DEPTH * 768 * DM * 2);
  ushort_t* bfXproj   = (ushort_t*)alloc((size_t)DEPTH * 44 * DI * 2);
  ushort_t* bfOutproj = (ushort_t*)alloc((size_t)DEPTH * DM * DI * 2);
  ushort_t* bfHead    = (ushort_t*)alloc((size_t)NCLS * DM * 2);
  ushort_t* bfA       = (ushort_t*)alloc((size_t)ML * KPATCH * 2);
  float*    resid     = (float*)alloc((size_t)ML * DM * 4);
  float*    hidden    = (float*)alloc((size_t)ML * DM * 4);
  ushort_t* hnbf      = (ushort_t*)alloc((size_t)ML * DM * 2);
  float*    xz        = (float*)alloc((size_t)ML * 768 * 4);
  ushort_t* xconvbf   = (ushort_t*)alloc((size_t)ML * DI * 2);
  float*    xdbl      = (float*)alloc((size_t)ML * 44 * 4);
  float*    dtbuf     = (float*)alloc((size_t)ML * DI * 4);
  ushort_t* ybf       = (ushort_t*)alloc((size_t)ML * DI * 2);
  ushort_t* pooledbf  = (ushort_t*)alloc((size_t)B_SZ * DM * 2);

  auto cvt = [&](const float* s, ushort_t* d, size_t n) {
    int blocks = (int)((n + 255) / 256); if (blocks > 8192) blocks = 8192;
    k_f32_to_bf16<<<blocks, 256, 0, stream>>>(s, d, (int)n);
  };
  cvt(patch_w,    bfPatch,   (size_t)DM * KPATCH);
  cvt(in_proj_w,  bfInproj,  (size_t)DEPTH * 768 * DM);
  cvt(x_proj_w,   bfXproj,   (size_t)DEPTH * 44 * DI);
  cvt(out_proj_w, bfOutproj, (size_t)DEPTH * DM * DI);
  cvt(head_w,     bfHead,    (size_t)NCLS * DM);

  k_im2col<<<(ML * KPATCH + 255) / 256, 256, 0, stream>>>(x, bfA);

  auto gemm = [&](const ushort_t* Am, const ushort_t* Wm, float* Cm,
                  int M, int N, int K, const float* bias) {
    dim3 g((N + 63) / 64, (M + 63) / 64);
    k_wmma_gemm<<<g, 128, 0, stream>>>(Am, Wm, Cm, M, N, K, bias);
  };

  // patch embedding -> hidden (B,L,192)
  gemm(bfA, bfPatch, hidden, ML, DM, KPATCH, patch_b);

  for (int i = 0; i < DEPTH; ++i) {
    k_resln<<<ML / 8, 256, 0, stream>>>(resid, hidden, (i == 0) ? 1 : 0,
                                        norm_w + (size_t)i * DM, norm_b + (size_t)i * DM, hnbf);
    gemm(hnbf, bfInproj + (size_t)i * 768 * DM, xz, ML, 768, DM, nullptr);
    k_conv_silu<<<(ML * DI + 255) / 256, 256, 0, stream>>>(
        xz, conv_w + (size_t)i * DI * DCONV, conv_b + (size_t)i * DI, xconvbf);
    gemm(xconvbf, bfXproj + (size_t)i * 44 * DI, xdbl, ML, 44, DI, nullptr);
    k_dtproj<<<(ML * DI + 255) / 256, 256, 0, stream>>>(
        xdbl, dt_proj_w + (size_t)i * DI * DTR, dt_proj_b + (size_t)i * DI, dtbuf);
    k_scan<<<B_SZ, DI, 0, stream>>>(xconvbf, dtbuf, xdbl,
                                    A_log + (size_t)i * DI * NSTATE,
                                    D_skip + (size_t)i * DI, xz, ybf);
    gemm(ybf, bfOutproj + (size_t)i * DM * DI, hidden, ML, DM, DI, nullptr);
  }

  k_resln<<<ML / 8, 256, 0, stream>>>(resid, hidden, 0, normf_w, normf_b, hnbf);
  k_pool<<<(B_SZ * DM + 255) / 256, 256, 0, stream>>>(hnbf, pooledbf);
  gemm(pooledbf, bfHead, out, B_SZ, NCLS, DM, head_b);
}
